// Feature_Correlation_Matching_90314572300426
// MI455X (gfx1250) — matrix operand order repaired and checked by reading, not compiled, not run on hardware
//
#include <hip/hip_runtime.h>
#include <hip/hip_bf16.h>
#include <math.h>

// ---------------------------------------------------------------------------
// Feature Correlation Matching on gfx1250 (MI455X)
//   x: (4, 64, 64, 256) f32 -> per batch: F = 4096 x 256
//   G = F F^T via v_wmma_f32_16x16x32_bf16, D_ij = ni + nj - 2 G_ij
//   fused branchless top-3 smallest per row (packed u64 keys), outputs:
//     pred  (4*4096 f32)  and  arg = index of 2nd-smallest (4*4096 i32)
// ---------------------------------------------------------------------------

typedef __attribute__((ext_vector_type(16))) __bf16 bf16x16;
typedef __attribute__((ext_vector_type(8)))  __bf16 bf16x8;
typedef __attribute__((ext_vector_type(8)))  float  f32x8;
typedef unsigned long long u64;

#define NB   4
#define NROW 4096
#define NC   256
#define KMAX 0xFFFFFFFFFFFFFFFFull

// Positive-float bit pattern is monotone as unsigned -> (dist, idx) lexicographic key.
__device__ __forceinline__ u64 pack_key(float d, int j) {
    return ((u64)__float_as_uint(d) << 32) | (unsigned int)j;
}

// Branchless sorted-insert of key k into ascending (b0 <= b1 <= b2).
__device__ __forceinline__ void top3_insert(u64& b0, u64& b1, u64& b2, u64 k) {
    const bool c0 = k < b0;
    const bool c1 = k < b1;
    const bool c2 = k < b2;
    const u64 nb2 = c1 ? b1 : (c2 ? k : b2);
    const u64 nb1 = c0 ? b0 : (c1 ? k : b1);
    const u64 nb0 = c0 ? k  : b0;
    b0 = nb0; b1 = nb1; b2 = nb2;
}

// ---------------------------------------------------------------------------
// Phase 1: convert f32 -> bf16 (row-major [b][i][k]) and compute per-row
// squared norms FROM THE ROUNDED bf16 VALUES (so D_ii ~ 0 like the reference).
// One wave of 32 lanes per row; each lane handles 8 contiguous elements.
// ---------------------------------------------------------------------------
__global__ void fcm_prep(const float* __restrict__ x,
                         __bf16* __restrict__ fmb,
                         float* __restrict__ norms) {
    const int row  = blockIdx.x * (blockDim.x >> 5) + (threadIdx.x >> 5);
    const int lane = threadIdx.x & 31;

    const float* src = x + (size_t)row * NC + lane * 8;
    float4 a = *(const float4*)(src);
    float4 b = *(const float4*)(src + 4);

    bf16x8 bv;
    bv[0] = (__bf16)a.x; bv[1] = (__bf16)a.y; bv[2] = (__bf16)a.z; bv[3] = (__bf16)a.w;
    bv[4] = (__bf16)b.x; bv[5] = (__bf16)b.y; bv[6] = (__bf16)b.z; bv[7] = (__bf16)b.w;

    float s = 0.0f;
#pragma unroll
    for (int i = 0; i < 8; ++i) { float r = (float)bv[i]; s = fmaf(r, r, s); }

    *(bf16x8*)(fmb + (size_t)row * NC + lane * 8) = bv;

    // wave32 butterfly reduction
#pragma unroll
    for (int off = 16; off > 0; off >>= 1) s += __shfl_xor(s, off, 32);
    if (lane == 0) norms[row] = s;
}

// ---------------------------------------------------------------------------
// Phase 2: one wave (32 threads) per 16-row tile. Sweep all 4096 columns in
// 16-wide tiles; 8x v_wmma_f32_16x16x32_bf16 per tile (K = 256). Branchless
// per-lane top-3 in packed u64 keys; merge via LDS at the end.
// ---------------------------------------------------------------------------
__global__ __launch_bounds__(32) void fcm_knn(const __bf16* __restrict__ fmb,
                                              const float* __restrict__ norms,
                                              float* __restrict__ pred_out,
                                              int* __restrict__ arg_out) {
    const int tile = blockIdx.x;          // 0..1023
    const int b    = tile >> 8;           // batch
    const int i0   = (tile & 255) << 4;   // row-tile base within batch
    const int lane = threadIdx.x;
    const int hb   = lane >> 4;           // lane half (K/M-half selector)
    const int ln   = lane & 15;           // column-within-tile / row-within-half

    const __bf16* fm = fmb + (size_t)b * NROW * NC;
    const float*  nr = norms + b * NROW;

    // --- A fragments for our 16 rows, 16-bit A 16x32 in contiguous-K order ---
    // lane ln holds row M=ln; hb=0 -> K [0..15]; hb=1 -> K [16..31] (contiguous, matches B)
    bf16x16 afrag[8];
    {
        const __bf16* arow = fm + (size_t)(i0 + ln) * NC;
#pragma unroll
        for (int s = 0; s < 8; ++s) {
            bf16x8 c0 = *(const bf16x8*)(arow + 32 * s +     16 * hb);
            bf16x8 c1 = *(const bf16x8*)(arow + 32 * s + 8 + 16 * hb);
            afrag[s] = __builtin_shufflevector(c0, c1,
                        0,1,2,3,4,5,6,7, 8,9,10,11,12,13,14,15);
        }
    }

    // norms of the 8 accumulator rows this lane owns (M = 8*hb + r)
    float rn[8];
#pragma unroll
    for (int r = 0; r < 8; ++r) rn[r] = nr[i0 + 8 * hb + r];

    // per-lane running top-3 as packed (dist_bits, idx) keys, ascending
    u64 k0[8], k1[8], k2[8];
#pragma unroll
    for (int r = 0; r < 8; ++r) { k0[r] = KMAX; k1[r] = KMAX; k2[r] = KMAX; }

    for (int j0 = 0; j0 < NROW; j0 += 16) {
        // --- B fragments: lane ln = column N, 16 contiguous K per lane ---
        const __bf16* brow = fm + (size_t)(j0 + ln) * NC + 16 * hb;
        bf16x16 bfrag[8];
#pragma unroll
        for (int s = 0; s < 8; ++s) bfrag[s] = *(const bf16x16*)(brow + 32 * s);

        f32x8 acc = {0.f,0.f,0.f,0.f,0.f,0.f,0.f,0.f};
#pragma unroll
        for (int s = 0; s < 8; ++s)
            acc = __builtin_amdgcn_wmma_f32_16x16x32_bf16(
                      false, afrag[s], false, bfrag[s],
                      (short)0, acc, false, false);

        const int   j  = j0 + ln;
        const float cn = nr[j];
#pragma unroll
        for (int r = 0; r < 8; ++r) {
            float d = fmaf(-2.0f, acc[r], rn[r] + cn);
            d = fmaxf(d, 0.0f) + 1e-9f;
            top3_insert(k0[r], k1[r], k2[r], pack_key(d, j));
        }
    }

    // --- merge per-lane candidates across the 16 lanes of each half via LDS ---
    __shared__ u64 sk[32][8][3];
#pragma unroll
    for (int r = 0; r < 8; ++r) {
        sk[lane][r][0] = k0[r];
        sk[lane][r][1] = k1[r];
        sk[lane][r][2] = k2[r];
    }
    __syncthreads();

    if (lane < 16) {
        const int m    = lane;            // row within tile
        const int half = m >> 3;          // which lane-half produced it
        const int r    = m & 7;           // accumulator register index
        u64 t0 = KMAX, t1 = KMAX, t2 = KMAX;
#pragma unroll
        for (int ll = 0; ll < 16; ++ll) {
#pragma unroll
            for (int c = 0; c < 3; ++c)
                top3_insert(t0, t1, t2, sk[half * 16 + ll][r][c]);
        }
        const float d1 = sqrtf(__uint_as_float((unsigned int)(t1 >> 32)));
        const float d2 = sqrtf(__uint_as_float((unsigned int)(t2 >> 32)));
        const float e  = expf(d1);
        const float pred = (d1 / d2 < 0.6f) ? 2.0f / (1.0f + e)
                                            : 2.0f / (1.0f + 2.0f * e);
        const int gi = b * NROW + i0 + m;
        pred_out[gi] = pred;
        arg_out[gi]  = (int)(unsigned int)(t1 & 0xFFFFFFFFull);
    }
}

// ---------------------------------------------------------------------------
extern "C" void kernel_launch(void* const* d_in, const int* in_sizes, int n_in,
                              void* d_out, int out_size, void* d_ws, size_t ws_size,
                              hipStream_t stream) {
    const float* x = (const float*)d_in[0];

    __bf16* fmb   = (__bf16*)d_ws;                                   // 8 MB
    float*  norms = (float*)((char*)d_ws + (size_t)NB * NROW * NC * sizeof(__bf16));

    float* pred = (float*)d_out;                     // first 16384 floats
    int*   arg  = (int*)((float*)d_out + NB * NROW); // next 16384 as int32

    // Phase 1: 16384 rows, 8 waves (256 threads) per block
    fcm_prep<<<(NB * NROW) / 8, 256, 0, stream>>>(x, fmb, norms);

    // Phase 2: one wave per 16-row tile: 4 batches * 256 tiles = 1024 blocks
    fcm_knn<<<NB * (NROW / 16), 32, 0, stream>>>(fmb, norms, pred, arg);
}
